// Physics_Attention_Irregular_Mesh_17068200034964
// MI455X (gfx1250) — matrix operand order repaired and checked by reading, not compiled, not run on hardware
//
#include <hip/hip_runtime.h>
#include <math.h>

// ---------------- problem constants ----------------
constexpr int BB    = 4;
constexpr int NN    = 32768;
constexpr int DIMX  = 256;
constexpr int HH    = 8;
constexpr int DD    = 64;
constexpr int GG    = 64;
constexpr int INNERX = HH * DD;   // 512
constexpr float EPS_SLICE = 1e-5f;
constexpr float EPS_NORM  = 1e-12f;

// ---------------- WMMA fragment types --------------
typedef __attribute__((ext_vector_type(16))) __bf16 bf16x16;
typedef __attribute__((ext_vector_type(8)))  __bf16 bf16x8;
typedef __attribute__((ext_vector_type(4)))  __bf16 bf16x4;
typedef __attribute__((ext_vector_type(8)))  float  f32x8;

__device__ __forceinline__ f32x8 wmma_bf16(bf16x16 a, bf16x16 b, f32x8 c) {
  // D = A(16x32) * B(32x16) + C(16x16 f32)
  return __builtin_amdgcn_wmma_f32_16x16x32_bf16(
      false, a, false, b, (short)0, c, false, false);
}

// A fragment (16x32 bf16) from row-major src (row length = ld elements).
// Contiguous K order used here (B matches): row r=lane&15, K in {16h..16h+15}, h=lane>>4
__device__ __forceinline__ bf16x16 load_a_rm(const __bf16* base, int ld) {
  int lane = threadIdx.x & 31;
  int half = lane >> 4, r = lane & 15;
  const __bf16* p = base + (size_t)r * ld + 16 * half;
  bf16x8 lo = *(const bf16x8*)p;
  bf16x8 hi = *(const bf16x8*)(p + 8);
  return __builtin_shufflevector(lo, hi, 0, 1, 2, 3, 4, 5, 6, 7,
                                 8, 9, 10, 11, 12, 13, 14, 15);
}

// B fragment (32x16 bf16) where B = W^T: W row-major [Nout][K].
// lane holds output column j = jbase + (lane&15), K = kbase + 16*(lane>>4) + 0..15
// The 16 K-values are contiguous -> single 32-byte vector load.
__device__ __forceinline__ bf16x16 load_b_wt(const __bf16* W, int ldw,
                                             int jbase, int kbase) {
  int lane = threadIdx.x & 31;
  int khalf = lane >> 4, n = lane & 15;
  const __bf16* p = W + (size_t)(jbase + n) * ldw + kbase + 16 * khalf;
  return *(const bf16x16*)p;
}

__device__ __forceinline__ f32x8 zero8() {
  f32x8 z = {0.f, 0.f, 0.f, 0.f, 0.f, 0.f, 0.f, 0.f};
  return z;
}

// ---------------- small utility kernels ----------------
__global__ void k_zero_f32(float* p, long n) {
  long i = (long)blockIdx.x * blockDim.x + threadIdx.x;
  long s = (long)gridDim.x * blockDim.x;
  for (; i < n; i += s) p[i] = 0.f;
}

__global__ void k_cvt_bf16(__bf16* dst, const float* src, long n) {
  long i = (long)blockIdx.x * blockDim.x + threadIdx.x;
  long s = (long)gridDim.x * blockDim.x;
  for (; i < n; i += s) dst[i] = (__bf16)src[i];
}

// Wcomb[h*64+g][k] = (1/temp[h]) * sum_d Wslice[g,d] * Wx[h*64+d, k]
// bcomb[h*64+g]    = (1/temp[h]) * (bslice[g] + sum_d Wslice[g,d]*bx[h*64+d])
__global__ void k_build_wcomb(const float* Wx, const float* bx,
                              const float* Wslice, const float* bslice,
                              const float* temp, __bf16* Wcomb, float* bcomb) {
  int idx = blockIdx.x * blockDim.x + threadIdx.x;
  if (idx >= INNERX * DIMX) return;
  int j = idx / DIMX, k = idx % DIMX;
  int h = j >> 6, g = j & 63;
  float invt = 1.0f / temp[h];
  float s = 0.f;
  for (int d = 0; d < DD; ++d)
    s += Wslice[g * DD + d] * Wx[(h * DD + d) * DIMX + k];
  Wcomb[idx] = (__bf16)(s * invt);
  if (k == 0) {
    float bs = bslice[g];
    for (int d = 0; d < DD; ++d) bs += Wslice[g * DD + d] * bx[h * DD + d];
    bcomb[j] = bs * invt;
  }
}

// ---------------- K1: fused projections + slice routing softmax ----------------
// grid = B*N/16 blocks, 256 threads (8 waves, wave = head).
// x is staged f32->bf16 into LDS once per block; all 8 waves share the A tile.
__global__ __launch_bounds__(256)
void k_proj_route(const float* x, const __bf16* Wfx_b, const float* bfx,
                  const __bf16* Wcomb_b, const float* bcomb,
                  __bf16* fxb, __bf16* swb, float* snorm) {
  int bid = blockIdx.x;
  int b  = bid / (NN / 16);
  int n0 = (bid % (NN / 16)) * 16;
  int tid = threadIdx.x, wave = tid >> 5, lane = tid & 31;
  int h = wave;
  __shared__ __bf16 xs[16][DIMX];     // 8 KB staged A tile (bf16)
  __shared__ float  lg[HH][16 * GG];  // per-wave logits [16 tokens][64 g]

  // cooperative stage: 16x256 f32 -> bf16 LDS (1024 float4 chunks)
  {
    const float4* src = (const float4*)(x + ((size_t)b * NN + n0) * DIMX);
#pragma unroll
    for (int c = tid; c < 1024; c += 256) {
      float4 v = src[c];
      int r = c >> 6, col = (c & 63) * 4;
      bf16x4 w;
      w[0] = (__bf16)v.x; w[1] = (__bf16)v.y;
      w[2] = (__bf16)v.z; w[3] = (__bf16)v.w;
      *(bf16x4*)&xs[r][col] = w;
    }
  }
  __syncthreads();

  f32x8 cf[4] = {zero8(), zero8(), zero8(), zero8()};
  f32x8 cl[4] = {zero8(), zero8(), zero8(), zero8()};
#pragma unroll
  for (int kc = 0; kc < DIMX / 32; ++kc) {
    bf16x16 a = load_a_rm(&xs[0][0] + kc * 32, DIMX);
#pragma unroll
    for (int t = 0; t < 4; ++t) {
      cf[t] = wmma_bf16(a, load_b_wt(Wfx_b,   DIMX, h * 64 + t * 16, kc * 32), cf[t]);
      cl[t] = wmma_bf16(a, load_b_wt(Wcomb_b, DIMX, h * 64 + t * 16, kc * 32), cl[t]);
    }
  }
  int ncol = lane & 15, mh = (lane >> 4) * 8;
  // fx (+bias) -> bf16 global
#pragma unroll
  for (int t = 0; t < 4; ++t) {
    int j = h * 64 + t * 16 + ncol;
    float bj = bfx[j];
#pragma unroll
    for (int i = 0; i < 8; ++i)
      fxb[((size_t)(b * NN + n0 + mh + i)) * INNERX + j] = (__bf16)(cf[t][i] + bj);
  }
  // logits (+bias) -> LDS
#pragma unroll
  for (int t = 0; t < 4; ++t) {
    int g = t * 16 + ncol;
    float bj = bcomb[h * 64 + g];
#pragma unroll
    for (int i = 0; i < 8; ++i)
      lg[wave][(mh + i) * GG + g] = cl[t][i] + bj;
  }
  __syncthreads();
  // softmax over g per token (lanes 0..15, one token each)
  if (lane < 16) {
    float* row = &lg[wave][lane * GG];
    float mx = -3.4e38f;
    for (int g = 0; g < GG; ++g) mx = fmaxf(mx, row[g]);
    float ssum = 0.f;
    for (int g = 0; g < GG; ++g) { float e = __expf(row[g] - mx); row[g] = e; ssum += e; }
    float inv = 1.0f / ssum;
    __bf16* dst = swb + ((size_t)(b * HH + h) * NN + n0 + lane) * GG;
#pragma unroll
    for (int c = 0; c < 8; ++c) {
      bf16x8 v;
#pragma unroll
      for (int i = 0; i < 8; ++i) {
        float w = row[c * 8 + i] * inv;
        row[c * 8 + i] = w;
        v[i] = (__bf16)w;
      }
      *(bf16x8*)(dst + c * 8) = v;
    }
  }
  __syncthreads();
  // snorm[b,h,g] += sum over 16 tokens
  for (int gi = lane; gi < GG; gi += 32) {
    float s = 0.f;
    for (int t = 0; t < 16; ++t) s += lg[wave][t * GG + gi];
    atomicAdd(&snorm[(b * HH + h) * GG + gi], s);
  }
}

// ---------------- K3: st[b,h,g,d] = sum_n sw[n,g]*fx[n,d] ----------------
// grid = B*H*NSLAB blocks, 256 threads. Streaming over N with prefetch.
constexpr int NSLAB = 8;
__global__ __launch_bounds__(256)
void k_st_gemm(const __bf16* swb, const __bf16* fxb, float* st) {
  const int TOK = NN / NSLAB;           // 4096 tokens per slab
  int bid = blockIdx.x;
  int b    = bid / (HH * NSLAB);
  int rem  = bid % (HH * NSLAB);
  int h    = rem / NSLAB;
  int slab = rem % NSLAB;
  int tid = threadIdx.x, wave = tid >> 5, lane = tid & 31;
  int gtile = wave >> 1;                // 0..3
  int dbase = (wave & 1) * 2;           // d-tiles {0,1} or {2,3}

  __shared__ __bf16 sw_s[32][GG];
  __shared__ __bf16 fx_s[32][DD];

  f32x8 acc0 = zero8(), acc1 = zero8();
  int row = tid >> 3, c8 = (tid & 7) * 8;
  size_t swbase = (size_t)(b * HH + h) * NN * GG;
  size_t fxbase = (size_t)b * NN * INNERX + h * DD;

  for (int it = 0; it < TOK / 32; ++it) {
    int n0 = slab * TOK + it * 32;
    const __bf16* swp = swb + swbase + (size_t)(n0 + row) * GG + c8;
    const __bf16* fxp = fxb + fxbase + (size_t)(n0 + row) * INNERX + c8;
    *(bf16x8*)&sw_s[row][c8] = *(const bf16x8*)swp;
    *(bf16x8*)&fx_s[row][c8] = *(const bf16x8*)fxp;
    if (it + 1 < TOK / 32) {
      // pull next K-chunk toward GL2 while this one is consumed
      __builtin_prefetch(swp + 32 * GG, 0, 1);
      __builtin_prefetch(fxp + 32 * INNERX, 0, 1);
    }
    __syncthreads();
    // A = sw^T (g rows, n as K) built from LDS (transpose read)
    bf16x16 a;
    int g  = gtile * 16 + (lane & 15);
    int kh = (lane >> 4) * 16;
#pragma unroll
    for (int e = 0; e < 8; ++e) { a[e] = sw_s[kh + e][g]; a[e + 8] = sw_s[kh + 8 + e][g]; }
    // B = fx (n as K, d cols)
    int nc = lane & 15;
    int kb = (lane >> 4) * 16;
    bf16x16 b0, b1;
#pragma unroll
    for (int e = 0; e < 16; ++e) {
      b0[e] = fx_s[kb + e][dbase * 16 + nc];
      b1[e] = fx_s[kb + e][(dbase + 1) * 16 + nc];
    }
    acc0 = wmma_bf16(a, b0, acc0);
    acc1 = wmma_bf16(a, b1, acc1);
    __syncthreads();
  }
  int nc = lane & 15, mh = (lane >> 4) * 8;
#pragma unroll
  for (int i = 0; i < 8; ++i) {
    int g = gtile * 16 + mh + i;
    float* dst = &st[(((size_t)(b * HH + h) * GG + g)) * DD];
    atomicAdd(dst + dbase * 16 + nc,       acc0[i]);
    atomicAdd(dst + (dbase + 1) * 16 + nc, acc1[i]);
  }
}

// ---------------- K4: tiny slice-token attention (f32) ----------------
__global__ void k4_stn(const float* st, const float* snorm, float* stn, int n) {
  int i = blockIdx.x * blockDim.x + threadIdx.x;
  if (i < n) stn[i] = st[i] / (snorm[i >> 6] + EPS_SLICE);
}
__global__ void k4_kv(const float* stn, float* kv, int n) {
  int i = blockIdx.x * blockDim.x + threadIdx.x;
  if (i >= n) return;
  int b = i / (GG * DD), gd = i % (GG * DD);
  float s = 0.f;
  for (int h = 0; h < HH; ++h) s += stn[((size_t)(b * HH + h)) * GG * DD + gd];
  kv[i] = s * (1.0f / HH);
}
__global__ void k4_kvproj(const float* kv, const float* Wk, const float* Wv,
                          float* kf, float* vf, int n) {
  int i = blockIdx.x * blockDim.x + threadIdx.x;
  if (i >= n) return;
  int d = i & 63;
  const float* r = kv + (i & ~63);
  float sk = 0.f, sv = 0.f;
  for (int c = 0; c < DD; ++c) { float x = r[c]; sk += x * Wk[d * DD + c]; sv += x * Wv[d * DD + c]; }
  kf[i] = sk; vf[i] = sv;
}
__global__ void k4_rownorm(const float* src, float* dst, int rows) {
  int i = blockIdx.x * blockDim.x + threadIdx.x;
  if (i >= rows) return;
  const float* r = src + (size_t)i * DD;
  float s = 0.f;
  for (int c = 0; c < DD; ++c) s += r[c] * r[c];
  float inv = 1.0f / fmaxf(sqrtf(s), EPS_NORM);
  for (int c = 0; c < DD; ++c) dst[(size_t)i * DD + c] = r[c] * inv;
}
__global__ void k4_q(const float* stn, const float* Wq, float* q, int n) {
  int i = blockIdx.x * blockDim.x + threadIdx.x;
  if (i >= n) return;
  int d = i & 63;
  const float* r = stn + (i & ~63);
  float s = 0.f;
  for (int c = 0; c < DD; ++c) s += r[c] * Wq[d * DD + c];
  q[i] = s;
}
// grid = B*H*G blocks of 64 threads; writes out_st TRANSPOSED [b,h,d,g] in bf16
__global__ __launch_bounds__(64)
void k4_attn(const float* qn, const float* kn, const float* vf, const float* stn,
             const float* attn_scale, const float* res_scale, __bf16* out_stT) {
  int bid = blockIdx.x;
  int b = bid / (HH * GG), rem = bid % (HH * GG), h = rem / GG, g = rem % GG;
  int t = threadIdx.x;   // 0..63
  __shared__ float sl[GG], se[GG], sa[GG];
  const float* qrow = qn + ((size_t)(b * HH + h) * GG + g) * DD;
  float acc = 0.f;
  for (int c = 0; c < DD; ++c) acc += qrow[c] * kn[(size_t)(b * GG + t) * DD + c];
  sl[t] = acc * attn_scale[h];
  __syncthreads();
  float mx = -3.4e38f;
  for (int s = 0; s < GG; ++s) mx = fmaxf(mx, sl[s]);
  float e = __expf(sl[t] - mx);
  se[t] = e;
  __syncthreads();
  float sum = 0.f;
  for (int s = 0; s < GG; ++s) sum += se[s];
  sa[t] = e / sum;
  __syncthreads();
  int d = t;
  float o = 0.f;
  for (int s = 0; s < GG; ++s) o += sa[s] * vf[(size_t)(b * GG + s) * DD + d];
  o += res_scale[0] * stn[((size_t)(b * HH + h) * GG + g) * DD + d];
  out_stT[((size_t)(b * HH + h) * DD + d) * GG + g] = (__bf16)o;
}

// ---------------- K5: scatter back through sw + final projection ----------------
// grid = B*N/16 blocks, 256 threads (8 waves)
__global__ __launch_bounds__(256)
void k_out(const __bf16* swb, const __bf16* out_stT, const __bf16* Wout_b,
           const float* bout, float* out) {
  int bid = blockIdx.x;
  int b  = bid / (NN / 16);
  int n0 = (bid % (NN / 16)) * 16;
  int tid = threadIdx.x, wave = tid >> 5, lane = tid & 31;
  __shared__ __bf16 oxs[16][INNERX];   // 16 tokens x 512 merged-head features

  // stage 1: ox[n, h*64+d] = sum_g sw[n,g] * out_st[h][g,d]   (wave = head)
  int h = wave;
  const __bf16* swA  = swb + ((size_t)(b * HH + h) * NN + n0) * GG;
  const __bf16* ostT = out_stT + (size_t)(b * HH + h) * GG * DD;  // [d][g], ld=GG
  f32x8 cf[4] = {zero8(), zero8(), zero8(), zero8()};
#pragma unroll
  for (int kc = 0; kc < GG / 32; ++kc) {
    bf16x16 a = load_a_rm(swA + kc * 32, GG);
#pragma unroll
    for (int t = 0; t < 4; ++t)
      cf[t] = wmma_bf16(a, load_b_wt(ostT, GG, t * 16, kc * 32), cf[t]);
  }
  int ncol = lane & 15, mh = (lane >> 4) * 8;
#pragma unroll
  for (int t = 0; t < 4; ++t)
#pragma unroll
    for (int i = 0; i < 8; ++i)
      oxs[mh + i][h * 64 + t * 16 + ncol] = (__bf16)cf[t][i];
  __syncthreads();

  // stage 2: out[n, j] = sum_k ox[n,k] * Wout[j,k] + bout[j]; wave owns 32 cols
  int j0 = wave * 32;
  f32x8 co[2] = {zero8(), zero8()};
#pragma unroll
  for (int kc = 0; kc < INNERX / 32; ++kc) {
    bf16x16 a = load_a_rm(&oxs[0][0] + kc * 32, INNERX);
    co[0] = wmma_bf16(a, load_b_wt(Wout_b, INNERX, j0,      kc * 32), co[0]);
    co[1] = wmma_bf16(a, load_b_wt(Wout_b, INNERX, j0 + 16, kc * 32), co[1]);
  }
#pragma unroll
  for (int u = 0; u < 2; ++u) {
    int j = j0 + u * 16 + ncol;
    float bj = bout[j];
#pragma unroll
    for (int i = 0; i < 8; ++i)
      out[((size_t)(b * NN + n0 + mh + i)) * DIMX + j] = co[u][i] + bj;
  }
}

// ---------------- host launcher ----------------
extern "C" void kernel_launch(void* const* d_in, const int* in_sizes, int n_in,
                              void* d_out, int out_size, void* d_ws, size_t ws_size,
                              hipStream_t stream) {
  const float* x        = (const float*)d_in[0];
  const float* Wfx      = (const float*)d_in[1];
  const float* bfx      = (const float*)d_in[2];
  const float* Wx       = (const float*)d_in[3];
  const float* bx       = (const float*)d_in[4];
  const float* Wslice   = (const float*)d_in[5];
  const float* bslice   = (const float*)d_in[6];
  const float* temp     = (const float*)d_in[7];
  const float* Wq       = (const float*)d_in[8];
  const float* Wk       = (const float*)d_in[9];
  const float* Wv       = (const float*)d_in[10];
  const float* res_scale  = (const float*)d_in[11];
  const float* attn_scale = (const float*)d_in[12];
  const float* Wout     = (const float*)d_in[13];
  const float* bout     = (const float*)d_in[14];
  float* out = (float*)d_out;

  // workspace carve-out
  char* wp = (char*)d_ws;
  auto carve = [&](size_t bytes) {
    void* p = (void*)wp;
    wp += (bytes + 255) & ~(size_t)255;
    return p;
  };
  float*  snorm   = (float*)carve((size_t)BB * HH * GG * 4);
  float*  st      = (float*)carve((size_t)BB * HH * GG * DD * 4);
  float*  stn     = (float*)carve((size_t)BB * HH * GG * DD * 4);
  float*  kv      = (float*)carve((size_t)BB * GG * DD * 4);
  float*  kf      = (float*)carve((size_t)BB * GG * DD * 4);
  float*  vf      = (float*)carve((size_t)BB * GG * DD * 4);
  float*  kn      = (float*)carve((size_t)BB * GG * DD * 4);
  float*  qv      = (float*)carve((size_t)BB * HH * GG * DD * 4);
  float*  qn      = (float*)carve((size_t)BB * HH * GG * DD * 4);
  float*  bcomb   = (float*)carve((size_t)INNERX * 4);
  __bf16* Wfx_b   = (__bf16*)carve((size_t)INNERX * DIMX * 2);
  __bf16* Wcomb_b = (__bf16*)carve((size_t)INNERX * DIMX * 2);
  __bf16* Wout_b  = (__bf16*)carve((size_t)DIMX * INNERX * 2);
  __bf16* ostT    = (__bf16*)carve((size_t)BB * HH * GG * DD * 2);
  __bf16* fxb     = (__bf16*)carve((size_t)BB * NN * INNERX * 2);
  __bf16* swb     = (__bf16*)carve((size_t)BB * HH * NN * GG * 2);

  // 1) zero atomic accumulators
  k_zero_f32<<<8,   256, 0, stream>>>(snorm, (long)BB * HH * GG);
  k_zero_f32<<<512, 256, 0, stream>>>(st,    (long)BB * HH * GG * DD);

  // 2) bf16 weight conversions (x converts in-kernel during LDS staging)
  k_cvt_bf16<<<512, 256, 0, stream>>>(Wfx_b,  Wfx,  (long)INNERX * DIMX);
  k_cvt_bf16<<<512, 256, 0, stream>>>(Wout_b, Wout, (long)DIMX * INNERX);

  // 3) combined routing weight (folds Wx, Wslice, 1/temp)
  k_build_wcomb<<<(INNERX * DIMX + 255) / 256, 256, 0, stream>>>(
      Wx, bx, Wslice, bslice, temp, Wcomb_b, bcomb);

  // 4) fused projection + routing softmax
  k_proj_route<<<BB * NN / 16, 256, 0, stream>>>(
      x, Wfx_b, bfx, Wcomb_b, bcomb, fxb, swb, snorm);

  // 5) slice-token pooling GEMM (K = N, split over 8 slabs, atomic reduce)
  k_st_gemm<<<BB * HH * NSLAB, 256, 0, stream>>>(swb, fxb, st);

  // 6) tiny slice-token attention (f32)
  int nSt = BB * HH * GG * DD;     // 131072
  int nKv = BB * GG * DD;          // 16384
  k4_stn    <<<(nSt + 255) / 256, 256, 0, stream>>>(st, snorm, stn, nSt);
  k4_kv     <<<(nKv + 255) / 256, 256, 0, stream>>>(stn, kv, nKv);
  k4_kvproj <<<(nKv + 255) / 256, 256, 0, stream>>>(kv, Wk, Wv, kf, vf, nKv);
  k4_rownorm<<<(BB * GG + 255) / 256, 256, 0, stream>>>(kf, kn, BB * GG);
  k4_q      <<<(nSt + 255) / 256, 256, 0, stream>>>(stn, Wq, qv, nSt);
  k4_rownorm<<<(BB * HH * GG + 255) / 256, 256, 0, stream>>>(qv, qn, BB * HH * GG);
  k4_attn   <<<BB * HH * GG, 64, 0, stream>>>(qn, kn, vf, stn,
                                              attn_scale, res_scale, ostT);

  // 7) scatter back + output projection
  k_out<<<BB * NN / 16, 256, 0, stream>>>(swb, ostT, Wout_b, bout, out);

  (void)in_sizes; (void)n_in; (void)out_size; (void)ws_size;
}
